// CustomGNN_14602888806632
// MI455X (gfx1250) — matrix operand order repaired and checked by reading, not compiled, not run on hardware
//
#include <hip/hip_runtime.h>

typedef __attribute__((ext_vector_type(16))) _Float16 v16h;
typedef __attribute__((ext_vector_type(8)))  _Float16 v8h;
typedef __attribute__((ext_vector_type(8)))  float    v8f;

// ---------------------------------------------------------------------------
// WMMA GEMM: out[M,Nout] = Ah[M,Kp](f16) @ W(col-major f16 [Nout,Kp]) + bias
// One wave computes 5 consecutive 16-row tiles x one 16-col tile.
// 4 waves / block handle 4 different column tiles.
// Fragment layouts used here, k contiguous in A and B (16-bit A 16x32 / B 32x16, wave32):
//   A: lane L row M=L&15; lanes 0-15 elems = K k0+0..15;
//                          lanes16-31 elems = K k0+16..31.
//   B: lane L col N=L&15; lanes 0-15 elems = K k0+0..15; lanes16-31 K k0+16..31.
//   C/D: lane L col N=L&15; VGPR r -> row M = 8*(L>>4)+r.
// All loads are unconditional 16B global_load_b128 (Kp % 32 == 0, zero-padded).
// ---------------------------------------------------------------------------
template<int K>
__global__ __launch_bounds__(128)
void gemm_wmma_f16(const _Float16* __restrict__ Ah,  // [M, K] row-major f16 (padded)
                   const _Float16* __restrict__ Wt,  // [Nout, K] f16 (transposed W)
                   const float* __restrict__ bias,   // [Nout]
                   float* __restrict__ out,          // [M, Nout] f32
                   int M, int Nout)
{
    constexpr int MT = 5;                       // 5 M-tiles per wave (80 rows)
    const int lane = threadIdx.x & 31;
    const int wave = threadIdx.x >> 5;
    const int tm0  = blockIdx.x * (16 * MT);
    const int tn   = (blockIdx.y * 4 + wave) * 16;

    const int half = lane >> 4;
    const int lr   = lane & 15;
    const int col  = tn + lr;

    const _Float16* __restrict__ wcol = Wt + (size_t)col * K;

    v8f acc[MT];
#pragma unroll
    for (int m = 0; m < MT; ++m) acc[m] = (v8f){};

#pragma unroll
    for (int k0 = 0; k0 < K; k0 += 32) {
        const int kb = k0 + half * 16;
        const v8h b0 = *(const v8h*)(wcol + kb);
        const v8h b1 = *(const v8h*)(wcol + kb + 8);
        const v16h b = __builtin_shufflevector(b0, b1,
                          0,1,2,3,4,5,6,7,8,9,10,11,12,13,14,15);
        const int ka = k0 + half * 16;
#pragma unroll
        for (int m = 0; m < MT; ++m) {
            const _Float16* __restrict__ arow = Ah + (size_t)(tm0 + 16*m + lr) * K;
            const v8h a0 = *(const v8h*)(arow + ka);
            const v8h a1 = *(const v8h*)(arow + ka + 8);
            const v16h a = __builtin_shufflevector(a0, a1,
                              0,1,2,3,4,5,6,7,8,9,10,11,12,13,14,15);
            acc[m] = __builtin_amdgcn_wmma_f32_16x16x32_f16(
                         false, a, false, b, (short)0, acc[m], false, false);
        }
    }

    const float bv = bias[col];
#pragma unroll
    for (int m = 0; m < MT; ++m) {
#pragma unroll
        for (int r = 0; r < 8; ++r) {
            out[(size_t)(tm0 + 16*m + half*8 + r) * Nout + col] = acc[m][r] + bv;
        }
    }
}

// ---------------------- prep: f32 -> padded f16 conversions -----------------
__global__ void cvt_pad_f16_kernel(const float* __restrict__ in, _Float16* __restrict__ outh,
                                   int M, int Kin, int Kp)
{
    int i = blockIdx.x * blockDim.x + threadIdx.x;
    if (i >= M * Kp) return;
    int r = i / Kp, k = i - r * Kp;
    outh[i] = (k < Kin) ? (_Float16)in[(size_t)r * Kin + k] : (_Float16)0.0f;
}

// W [Kin, Nout] f32 row-major  ->  Wt [Nout, Kp] f16 (column of W contiguous)
__global__ void wt_cvt_kernel(const float* __restrict__ W, _Float16* __restrict__ Wt,
                              int Kin, int Kp, int Nout)
{
    int i = blockIdx.x * blockDim.x + threadIdx.x;
    if (i >= Nout * Kp) return;
    int c = i / Kp, k = i - c * Kp;
    Wt[i] = (k < Kin) ? (_Float16)W[(size_t)k * Nout + c] : (_Float16)0.0f;
}

// ------------------------- float <-> ordered-int (atomicMax on float) -------
__device__ __forceinline__ int f2ord(float f) {
    int i = __float_as_int(f);
    return (i >= 0) ? i : (i ^ 0x7fffffff);
}
__device__ __forceinline__ float ord2f(int k) {
    return __int_as_float((k >= 0) ? k : (k ^ 0x7fffffff));
}

// ------------------------- per-layer helper kernels -------------------------
__global__ void init_nodes_kernel(int* __restrict__ maxb, float* __restrict__ den, int n)
{
    int i = blockIdx.x * blockDim.x + threadIdx.x;
    if (i < n) { maxb[i] = INT_MIN; den[i] = 0.0f; }
}

__global__ void init_out_kernel(float* __restrict__ out, const float* __restrict__ b,
                                int N, int HC)
{
    int i = blockIdx.x * blockDim.x + threadIdx.x;
    if (i < N * HC) out[i] = b[i % HC];
}

// logit[e,h] = att[h,:] . leaky_relu(xl[src]+xr[dst]+ea*We[h,:]) ; segment max
// C is compile-time so every gather becomes global_load_b128 (bases 16B aligned).
template<int C>
__global__ void edge_logit_kernel(const float* __restrict__ xl, const float* __restrict__ xr,
                                  const float* __restrict__ ea, const float* __restrict__ We,
                                  const float* __restrict__ att,
                                  const int* __restrict__ src, const int* __restrict__ dst,
                                  float* __restrict__ logit, int* __restrict__ maxb, int E)
{
    int t = blockIdx.x * blockDim.x + threadIdx.x;
    if (t >= E * 8) return;
    const int e = t >> 3, h = t & 7;
    const int s = src[e], d = dst[e];
    const float eav = ea[e];
    constexpr int HC = 8 * C;
    const float4* __restrict__ pl = (const float4*)(xl + (size_t)s * HC + h * C);
    const float4* __restrict__ pr = (const float4*)(xr + (size_t)d * HC + h * C);
    const float4* __restrict__ pw = (const float4*)(We + h * C);
    const float4* __restrict__ pa = (const float4*)(att + h * C);
    float acc = 0.0f;
#pragma unroll
    for (int q = 0; q < C / 4; ++q) {
        const float4 a = pl[q], r = pr[q], w = pw[q], at = pa[q];
        float m0 = a.x + r.x + eav * w.x;  m0 = (m0 > 0.0f) ? m0 : 0.2f * m0;
        float m1 = a.y + r.y + eav * w.y;  m1 = (m1 > 0.0f) ? m1 : 0.2f * m1;
        float m2 = a.z + r.z + eav * w.z;  m2 = (m2 > 0.0f) ? m2 : 0.2f * m2;
        float m3 = a.w + r.w + eav * w.w;  m3 = (m3 > 0.0f) ? m3 : 0.2f * m3;
        acc += m0 * at.x + m1 * at.y + m2 * at.z + m3 * at.w;
    }
    logit[t] = acc;
    atomicMax(&maxb[d * 8 + h], f2ord(acc));
}

// ex = exp(logit - max[dst]) ; segment sum of ex
__global__ void edge_exp_kernel(float* __restrict__ logit, const int* __restrict__ maxb,
                                float* __restrict__ den, const int* __restrict__ dst, int E)
{
    int t = blockIdx.x * blockDim.x + threadIdx.x;
    if (t >= E * 8) return;
    const int e = t >> 3, h = t & 7;
    const int d = dst[e];
    const float mx = ord2f(maxb[d * 8 + h]);
    const float ex = __expf(logit[t] - mx);
    logit[t] = ex;
    atomicAdd(&den[d * 8 + h], ex);
}

// out[dst] += (ex/den) * xl[src]   (b128 gathers of xl, per-dword atomic scatter)
template<int C>
__global__ void edge_aggregate_kernel(const float* __restrict__ xl,
                                      const float* __restrict__ logit,
                                      const float* __restrict__ den,
                                      const int* __restrict__ src, const int* __restrict__ dst,
                                      float* __restrict__ out, int E)
{
    int t = blockIdx.x * blockDim.x + threadIdx.x;
    if (t >= E * 8) return;
    const int e = t >> 3, h = t & 7;
    const int s = src[e], d = dst[e];
    const float alpha = logit[t] / (den[d * 8 + h] + 1e-16f);
    constexpr int HC = 8 * C;
    const float4* __restrict__ pl = (const float4*)(xl + (size_t)s * HC + h * C);
    float* __restrict__ po = out + (size_t)d * HC + h * C;
#pragma unroll
    for (int q = 0; q < C / 4; ++q) {
        const float4 a = pl[q];
        atomicAdd(&po[4 * q + 0], alpha * a.x);
        atomicAdd(&po[4 * q + 1], alpha * a.y);
        atomicAdd(&po[4 * q + 2], alpha * a.z);
        atomicAdd(&po[4 * q + 3], alpha * a.w);
    }
}

__global__ void elu_kernel(float* __restrict__ x, int n)
{
    int i = blockIdx.x * blockDim.x + threadIdx.x;
    if (i < n) {
        float v = x[i];
        x[i] = (v > 0.0f) ? v : (__expf(v) - 1.0f);
    }
}

// final 64 -> 1 linear
__global__ void final_linear_kernel(const float* __restrict__ h,
                                    const float* __restrict__ Wlin,
                                    const float* __restrict__ blin,
                                    float* __restrict__ out, int N)
{
    int n = blockIdx.x * blockDim.x + threadIdx.x;
    if (n >= N) return;
    const float4* __restrict__ p = (const float4*)(h + (size_t)n * 64);
    const float4* __restrict__ wv = (const float4*)Wlin;
    float acc = blin[0];
#pragma unroll
    for (int j = 0; j < 16; ++j) {
        const float4 a = p[j], b = wv[j];
        acc += a.x * b.x + a.y * b.y + a.z * b.z + a.w * b.w;
    }
    out[n] = acc;
}

// ---------------------------------------------------------------------------
extern "C" void kernel_launch(void* const* d_in, const int* in_sizes, int n_in,
                              void* d_out, int out_size, void* d_ws, size_t ws_size,
                              hipStream_t stream)
{
    const float* x   = (const float*)d_in[0];      // [N,15]
    const float* ea  = (const float*)d_in[1];      // [E,1]
    const int*  eidx = (const int*)  d_in[2];      // [2,E]
    const int N = in_sizes[0] / 15;
    const int E = in_sizes[2] / 2;
    const int* src = eidx;
    const int* dst = eidx + E;

    const float* Wl[4], *bl[4], *Wr[4], *br[4], *We[4], *att[4], *bb[4];
    for (int l = 0; l < 4; ++l) {
        const int p = 3 + 7 * l;
        Wl[l]  = (const float*)d_in[p + 0];
        bl[l]  = (const float*)d_in[p + 1];
        Wr[l]  = (const float*)d_in[p + 2];
        br[l]  = (const float*)d_in[p + 3];
        We[l]  = (const float*)d_in[p + 4];
        att[l] = (const float*)d_in[p + 5];
        bb[l]  = (const float*)d_in[p + 6];
    }
    const float* Wlin = (const float*)d_in[31];
    const float* blin = (const float*)d_in[32];

    // ---- workspace carve ----
    char* w = (char*)d_ws;
    auto carve = [&](size_t bytes) {
        void* p = (void*)w;
        w += (bytes + 255) & ~(size_t)255;
        return p;
    };
    float*     nA    = (float*)    carve((size_t)N * 128 * sizeof(float));
    float*     nB    = (float*)    carve((size_t)N * 128 * sizeof(float));
    float*     nC    = (float*)    carve((size_t)N * 128 * sizeof(float));
    float*     logit = (float*)    carve((size_t)E * 8   * sizeof(float));
    int*       maxb  = (int*)      carve((size_t)N * 8   * sizeof(int));
    float*     den   = (float*)    carve((size_t)N * 8   * sizeof(float));
    _Float16*  Ah    = (_Float16*) carve((size_t)N * 128 * sizeof(_Float16));
    _Float16*  WtL   = (_Float16*) carve((size_t)128 * 128 * sizeof(_Float16));
    _Float16*  WtR   = (_Float16*) carve((size_t)128 * 128 * sizeof(_Float16));

    const int din[4] = {15, 128, 128, 128};
    const int Kpv[4] = {32, 128, 128, 128};    // padded K (multiple of 32)
    const int Cc [4] = {16, 16, 16, 8};
    const int HC [4] = {128, 128, 128, 64};

    const float* ins [4] = {x,  nB, nC, nB};
    float*       xls [4] = {nA, nA, nA, nA};
    float*       outs[4] = {nB, nC, nB, nC};   // also serves as xr before aggregate

    const int TB = 256;
    const int edgeBlocks = (E * 8 + TB - 1) / TB;
    const int nodeHBlk   = (N * 8 + TB - 1) / TB;

    for (int l = 0; l < 4; ++l) {
        const int K = din[l], Kp = Kpv[l], C = Cc[l], hc = HC[l];

        // prep: activations -> padded f16; weights -> transposed padded f16
        cvt_pad_f16_kernel<<<(N * Kp + TB - 1) / TB, TB, 0, stream>>>(ins[l], Ah, N, K, Kp);
        wt_cvt_kernel<<<(hc * Kp + TB - 1) / TB, TB, 0, stream>>>(Wl[l], WtL, K, Kp, hc);
        wt_cvt_kernel<<<(hc * Kp + TB - 1) / TB, TB, 0, stream>>>(Wr[l], WtR, K, Kp, hc);

        dim3 gGrid(N / 80, hc / 64);
        dim3 gBlk(128);
        if (Kp == 32) {
            gemm_wmma_f16<32><<<gGrid, gBlk, 0, stream>>>(Ah, WtL, bl[l], xls[l],  N, hc);
            gemm_wmma_f16<32><<<gGrid, gBlk, 0, stream>>>(Ah, WtR, br[l], outs[l], N, hc);
        } else {
            gemm_wmma_f16<128><<<gGrid, gBlk, 0, stream>>>(Ah, WtL, bl[l], xls[l],  N, hc);
            gemm_wmma_f16<128><<<gGrid, gBlk, 0, stream>>>(Ah, WtR, br[l], outs[l], N, hc);
        }

        init_nodes_kernel<<<nodeHBlk, TB, 0, stream>>>(maxb, den, N * 8);
        if (C == 16)
            edge_logit_kernel<16><<<edgeBlocks, TB, 0, stream>>>(xls[l], outs[l], ea, We[l],
                                                                 att[l], src, dst, logit, maxb, E);
        else
            edge_logit_kernel<8><<<edgeBlocks, TB, 0, stream>>>(xls[l], outs[l], ea, We[l],
                                                                att[l], src, dst, logit, maxb, E);
        edge_exp_kernel<<<edgeBlocks, TB, 0, stream>>>(logit, maxb, den, dst, E);
        // xr dead from here on -> reuse the buffer as the aggregation output
        init_out_kernel<<<(N * hc + TB - 1) / TB, TB, 0, stream>>>(outs[l], bb[l], N, hc);
        if (C == 16)
            edge_aggregate_kernel<16><<<edgeBlocks, TB, 0, stream>>>(xls[l], logit, den,
                                                                     src, dst, outs[l], E);
        else
            edge_aggregate_kernel<8><<<edgeBlocks, TB, 0, stream>>>(xls[l], logit, den,
                                                                    src, dst, outs[l], E);
        if (l < 3)
            elu_kernel<<<(N * hc + TB - 1) / TB, TB, 0, stream>>>(outs[l], N * hc);
    }

    final_linear_kernel<<<(N + TB - 1) / TB, TB, 0, stream>>>(outs[3], Wlin, blin,
                                                              (float*)d_out, N);
    (void)n_in; (void)out_size; (void)ws_size;
}
